// DeepSpeedMLP_24446953848859
// MI455X (gfx1250) — compile-verified, matrix operand order checked by reading, not run on hardware
//
#include <hip/hip_runtime.h>
#include <math.h>

// Problem constants (B=2, S=2048, H=4096, I=16384)
#define HDIM  4096
#define IDIM  16384
#define MROWS 4096   // B*S

typedef __attribute__((ext_vector_type(16))) __bf16 v16bf;
typedef __attribute__((ext_vector_type(8)))  __bf16 v8bf;
typedef __attribute__((ext_vector_type(8)))  float  v8f;
typedef int v4i_ld __attribute__((vector_size(16)));   // builtin's pointee type

// gfx1250 async global->LDS path (ASYNCcnt-tracked).
#if defined(__gfx1250__) && __has_builtin(__builtin_amdgcn_global_load_async_to_lds_b128) && \
    __has_builtin(__builtin_amdgcn_s_wait_asynccnt)
#define HAVE_ASYNC 1
#else
#define HAVE_ASYNC 0
#endif

#define TO_LDS(p) ((__attribute__((address_space(3))) v4i_ld*)(unsigned long long)(p))
#define TO_GBL(p) ((v4i_ld*)(p))

__device__ __forceinline__ unsigned short f2bf(float f) {
  union { float f; unsigned int u; } c; c.f = f;
  unsigned int r = c.u + 0x7FFFu + ((c.u >> 16) & 1u);   // round-to-nearest-even
  return (unsigned short)(r >> 16);
}

__device__ __forceinline__ float gelu_tanh(float x) {
  float x3 = x * x * x;
  return 0.5f * x * (1.0f + tanhf(0.7978845608028654f * (x + 0.044715f * x3)));
}

// ---------------------------------------------------------------------------
// Kernel 1: residual_add = input + residual + bias; LayerNorm over H.
// ---------------------------------------------------------------------------
__global__ __launch_bounds__(256)
void fused_bias_res_ln(const float* __restrict__ in,
                       const float* __restrict__ res,
                       const float* __restrict__ bias,
                       const float* __restrict__ nw,
                       const float* __restrict__ nb,
                       float* __restrict__ resadd,
                       unsigned short* __restrict__ lnbf) {
  const int row = blockIdx.x;
  const int tid = threadIdx.x;
  const size_t base = (size_t)row * HDIM;

  float vals[16];
  float s = 0.f, s2 = 0.f;
#pragma unroll
  for (int j = 0; j < 16; ++j) {
    const int h = tid + j * 256;
    const float v = in[base + h] + res[base + h] + bias[h];
    vals[j] = v; s += v; s2 += v * v;
  }

  __shared__ float r1[256];
  __shared__ float r2[256];
  r1[tid] = s; r2[tid] = s2;
  __syncthreads();
  for (int off = 128; off > 0; off >>= 1) {
    if (tid < off) { r1[tid] += r1[tid + off]; r2[tid] += r2[tid + off]; }
    __syncthreads();
  }
  const float mean = r1[0] * (1.0f / HDIM);
  const float var  = r2[0] * (1.0f / HDIM) - mean * mean;
  const float rstd = rsqrtf(var + 1e-12f);

#pragma unroll
  for (int j = 0; j < 16; ++j) {
    const int h = tid + j * 256;
    const float v = vals[j];
    resadd[base + h] = v;
    const float l = (v - mean) * rstd * nw[h] + nb[h];
    lnbf[base + h] = f2bf(l);
  }
}

// ---------------------------------------------------------------------------
// Kernel 2: fp32 -> bf16 weight conversion.
// ---------------------------------------------------------------------------
__global__ __launch_bounds__(256)
void cvt_f32_bf16(const float* __restrict__ src,
                  unsigned short* __restrict__ dst, long n) {
  const long i = ((long)blockIdx.x * blockDim.x + threadIdx.x) * 4;
  if (i + 3 < n) {
    const float4 v = *reinterpret_cast<const float4*>(src + i);
    ushort4 o;
    o.x = f2bf(v.x); o.y = f2bf(v.y); o.z = f2bf(v.z); o.w = f2bf(v.w);
    *reinterpret_cast<ushort4*>(dst + i) = o;
  }
}

// ---------------------------------------------------------------------------
// Kernel 3: NT GEMM  C[M,N] = A[M,K] * W[N,K]^T  via v_wmma_f32_16x16x32_bf16
// Block tile 128x128, K-tile 32, double-buffered async global->LDS staging.
// 8 waves; each wave owns a 32x64 tile = 2x4 WMMA accumulators.
//   EPI==0: C = bf16( gelu(acc + bias[n]) )
//   EPI==1: C = fp32( acc + resadd[m,n] + bias[n] )
// ---------------------------------------------------------------------------
#define BM 128
#define BN 128
#define BK 32
#define LDT 40               // BK + 8 halfword pad; 80 bytes/row keeps 16B align
#define TILE_USH (128 * LDT) // halfwords per tile buffer

template <int EPI>
__global__ __launch_bounds__(256)
void gemm_wmma(const unsigned short* __restrict__ A,    // [Mtot x Ktot] bf16
               const unsigned short* __restrict__ W,    // [Ntot x Ktot] bf16
               const float* __restrict__ bias,          // [Ntot]
               const float* __restrict__ resadd,        // [Mtot x Ntot] (EPI==1)
               unsigned short* __restrict__ Cb,         // bf16 out (EPI==0)
               float* __restrict__ Cf,                  // f32 out (EPI==1)
               int Mtot, int Ntot, int Ktot) {
  __shared__ unsigned short As[2 * TILE_USH];
  __shared__ unsigned short Ws[2 * TILE_USH];

  const int tid  = threadIdx.x;
  const int wid  = tid >> 5;          // wave32
  const int lane = tid & 31;
  const int wm   = wid & 3;           // 4 wave rows * 32 = 128
  const int wn   = wid >> 2;          // 2 wave cols * 64 = 128
  const int bm0  = blockIdx.y * BM;
  const int bn0  = blockIdx.x * BN;

  const int lh    = lane & 15;
  const int khalf = (lane >> 4) << 3; // ISA 16-bit A layout: K base 0 or 8

  const v8f vzero = {0.f, 0.f, 0.f, 0.f, 0.f, 0.f, 0.f, 0.f};
  v8f acc[2][4];
#pragma unroll
  for (int mi = 0; mi < 2; ++mi)
#pragma unroll
    for (int ni = 0; ni < 4; ++ni) acc[mi][ni] = vzero;

  // Stage one 128x32 A tile + 128x32 W tile into LDS buffer `buf`.
  // 512 16B-chunks per tile; 2 per thread per tile.
  auto stage = [&](int k0, int buf) {
    unsigned short* as = As + buf * TILE_USH;
    unsigned short* ws = Ws + buf * TILE_USH;
#pragma unroll
    for (int i = 0; i < 2; ++i) {
      const int c  = tid + i * 256;
      const int r  = c >> 2;          // 4 chunks per row (BK/8)
      const int kc = (c & 3) * 8;
#if HAVE_ASYNC
      __builtin_amdgcn_global_load_async_to_lds_b128(
          TO_GBL(&A[(size_t)(bm0 + r) * Ktot + k0 + kc]),
          TO_LDS(&as[r * LDT + kc]), 0, 0);
      __builtin_amdgcn_global_load_async_to_lds_b128(
          TO_GBL(&W[(size_t)(bn0 + r) * Ktot + k0 + kc]),
          TO_LDS(&ws[r * LDT + kc]), 0, 0);
#else
      const uint4 va =
          *reinterpret_cast<const uint4*>(&A[(size_t)(bm0 + r) * Ktot + k0 + kc]);
      *reinterpret_cast<uint4*>(&as[r * LDT + kc]) = va;
      const uint4 vw =
          *reinterpret_cast<const uint4*>(&W[(size_t)(bn0 + r) * Ktot + k0 + kc]);
      *reinterpret_cast<uint4*>(&ws[r * LDT + kc]) = vw;
#endif
    }
  };

  stage(0, 0);
  int buf = 0;
  for (int k0 = 0; k0 < Ktot; k0 += BK) {
#if HAVE_ASYNC
    __builtin_amdgcn_s_wait_asynccnt(0);   // my async writes for this stage done
#endif
    __syncthreads();                       // everyone's writes visible; also WAR
                                           // guard for buffer refilled below
    if (k0 + BK < Ktot) stage(k0 + BK, buf ^ 1);

    const unsigned short* as = As + buf * TILE_USH;
    const unsigned short* ws = Ws + buf * TILE_USH;

    v16bf af[2];
#pragma unroll
    for (int mi = 0; mi < 2; ++mi) {
      const int r = wm * 32 + mi * 16 + lh;
      const v8bf lo = *reinterpret_cast<const v8bf*>(&as[r * LDT + khalf]);
      const v8bf hi = *reinterpret_cast<const v8bf*>(&as[r * LDT + khalf + 16]);
      af[mi] = __builtin_shufflevector(lo, hi, 0, 1, 2, 3, 4, 5, 6, 7,
                                       8, 9, 10, 11, 12, 13, 14, 15);
    }
    v16bf wf[4];
#pragma unroll
    for (int ni = 0; ni < 4; ++ni) {
      const int r = wn * 64 + ni * 16 + lh;
      const v8bf lo = *reinterpret_cast<const v8bf*>(&ws[r * LDT + khalf]);
      const v8bf hi = *reinterpret_cast<const v8bf*>(&ws[r * LDT + khalf + 16]);
      wf[ni] = __builtin_shufflevector(lo, hi, 0, 1, 2, 3, 4, 5, 6, 7,
                                       8, 9, 10, 11, 12, 13, 14, 15);
    }
#pragma unroll
    for (int mi = 0; mi < 2; ++mi)
#pragma unroll
      for (int ni = 0; ni < 4; ++ni)
        acc[mi][ni] = __builtin_amdgcn_wmma_f32_16x16x32_bf16(
            false, af[mi], false, wf[ni], (short)0, acc[mi][ni], false, false);

    buf ^= 1;
  }

  // Epilogue: C layout => VGPR r: M = r + (lane<16 ? 0 : 8), N = lane&15.
  const int mhalf = (lane >> 4) << 3;
#pragma unroll
  for (int mi = 0; mi < 2; ++mi) {
#pragma unroll
    for (int ni = 0; ni < 4; ++ni) {
      const int mt = bm0 + wm * 32 + mi * 16 + mhalf;
      const int n  = bn0 + wn * 64 + ni * 16 + lh;
#pragma unroll
      for (int r = 0; r < 8; ++r) {
        const int m = mt + r;
        const float x = acc[mi][ni][r];
        if (EPI == 0) {
          const float g = gelu_tanh(x + bias[n]);
          Cb[(size_t)m * Ntot + n] = f2bf(g);
        } else {
          Cf[(size_t)m * Ntot + n] = x + resadd[(size_t)m * Ntot + n] + bias[n];
        }
      }
    }
  }
}

// ---------------------------------------------------------------------------
extern "C" void kernel_launch(void* const* d_in, const int* in_sizes, int n_in,
                              void* d_out, int out_size, void* d_ws, size_t ws_size,
                              hipStream_t stream) {
  const float* input    = (const float*)d_in[0];
  const float* residual = (const float*)d_in[1];
  const float* bias     = (const float*)d_in[2];
  const float* attn_nw  = (const float*)d_in[3];
  const float* attn_nb  = (const float*)d_in[4];
  const float* inter_w  = (const float*)d_in[5];   // [I, H]
  const float* inter_b  = (const float*)d_in[6];   // [I]
  const float* output_w = (const float*)d_in[7];   // [H, I]
  const float* output_b = (const float*)d_in[8];   // [H]
  float* out = (float*)d_out;

  // workspace layout
  char* ws = (char*)d_ws;
  float*          resadd  = (float*)(ws);                                //  64 MB
  unsigned short* lnbf    = (unsigned short*)(ws + ((size_t)64  << 20)); //  32 MB
  unsigned short* iwbf    = (unsigned short*)(ws + ((size_t)96  << 20)); // 128 MB
  unsigned short* owbf    = (unsigned short*)(ws + ((size_t)224 << 20)); // 128 MB
  unsigned short* interbf = (unsigned short*)(ws + ((size_t)352 << 20)); // 128 MB

  fused_bias_res_ln<<<MROWS, 256, 0, stream>>>(input, residual, bias,
                                               attn_nw, attn_nb, resadd, lnbf);

  const long nel = (long)IDIM * HDIM;
  const int cvtBlocks = (int)(nel / 4 / 256);
  cvt_f32_bf16<<<cvtBlocks, 256, 0, stream>>>(inter_w, iwbf, nel);
  cvt_f32_bf16<<<cvtBlocks, 256, 0, stream>>>(output_w, owbf, nel);

  dim3 g1(IDIM / BN, MROWS / BM);
  gemm_wmma<0><<<g1, 256, 0, stream>>>(lnbf, iwbf, inter_b, nullptr,
                                       interbf, nullptr, MROWS, IDIM, HDIM);

  dim3 g2(HDIM / BN, MROWS / BM);
  gemm_wmma<1><<<g2, 256, 0, stream>>>(interbf, owbf, output_b, resadd,
                                       nullptr, out, MROWS, HDIM, IDIM);
}
